// VisualSpeechQFormer_42863773614225
// MI455X (gfx1250) — compile-verified, matrix operand order checked by reading, not run on hardware
//
#include <hip/hip_runtime.h>
#include <hip/hip_bf16.h>

#define D_    1024
#define OUT_  4096
#define Q_    32
#define NL_   2
#define H_    8
#define HD_   128
#define W_    3
#define B_    8
#define T_    1024
#define L_    (Q_ + T_)
#define EPS_  1e-5f

typedef __attribute__((ext_vector_type(16))) __bf16        v16bf;
typedef __attribute__((ext_vector_type(8)))  float         v8f;
typedef __attribute__((ext_vector_type(4)))  unsigned int  u32x4;

union FragAB { v16bf v; u32x4 q[2]; };

#if defined(__AMDGCN__) && __has_builtin(__builtin_amdgcn_global_load_async_to_lds_b128)
#define ASYNC_OK 1
typedef int vi4_ __attribute__((vector_size(16)));   // matches builtin pointee
#define AS1 __attribute__((address_space(1)))
#define AS3 __attribute__((address_space(3)))
#else
#define ASYNC_OK 0
#endif

__device__ __forceinline__ unsigned short f2bf(float f) {
  unsigned u = __float_as_uint(f);
  u += 0x7FFFu + ((u >> 16) & 1u);          // round-to-nearest-even
  return (unsigned short)(u >> 16);
}

__device__ __forceinline__ float geluf(float x) {
  return 0.5f * x * (1.0f + erff(x * 0.70710678118654752440f));
}

// ---------------------------------------------------------------- elementwise
__global__ void k_cast_f32_bf16(const float* __restrict__ in,
                                unsigned short* __restrict__ out, long n) {
  long i = (long)blockIdx.x * blockDim.x + threadIdx.x;
  if (i < n) out[i] = f2bf(in[i]);
}

// ------------------------------------------------- conv1 (groups=8) + GELU
__global__ void k_conv1_gelu(const float* __restrict__ x,
                             const float* __restrict__ w,
                             const float* __restrict__ bias,
                             float* __restrict__ y) {
  long idx = (long)blockIdx.x * blockDim.x + threadIdx.x;
  if (idx >= (long)B_ * T_ * D_) return;
  int o  = (int)(idx & (D_ - 1));
  long r = idx >> 10;
  int t  = (int)(r & (T_ - 1));
  int b  = (int)(r >> 10);
  int g  = o >> 7;
  const float* wr = w + (long)o * (128 * W_);
  const float* xb = x + ((long)b * T_) * D_ + g * 128;
  float acc = bias[o];
  #pragma unroll
  for (int k = 0; k < W_; ++k) {
    int tt = t + k - 1;
    if (tt < 0 || tt >= T_) continue;
    const float* xr = xb + (long)tt * D_;
    for (int i = 0; i < 128; ++i) acc += wr[i * W_ + k] * xr[i];
  }
  y[idx] = geluf(acc);
}

// ----------------------------------------------- im2col for conv2 (bf16 out)
__global__ void k_im2col_bf16(const float* __restrict__ g1,
                              unsigned short* __restrict__ out) {
  long idx = (long)blockIdx.x * blockDim.x + threadIdx.x;
  if (idx >= (long)B_ * T_ * (3 * D_)) return;
  int col = (int)(idx % (3 * D_));
  long row = idx / (3 * D_);
  int k = col % 3, i = col / 3;
  int t = (int)(row % T_);
  int b = (int)(row / T_);
  int tt = t + k - 1;
  float v = (tt >= 0 && tt < T_) ? g1[((long)b * T_ + tt) * D_ + i] : 0.0f;
  out[idx] = f2bf(v);
}

// ------------------------------------------- bf16 WMMA GEMM, LDS block-tiled
// C = act(A[M,K] * Bw[N,K]^T + bias) (+resid), orow remap for conv2.
// Block: 256 thr = 8 waves (4x2), tile 128x128, BK=32, double-buffered LDS.
#define BM_  128
#define BN_  128
#define BK_  32
#define LDSS 40   /* shorts per LDS tile row (80B, padded vs 64B) */

__global__ void __launch_bounds__(256)
k_gemm_bf16(const unsigned short* __restrict__ A,
            const unsigned short* __restrict__ Bw,
            const float* __restrict__ bias,
            void* __restrict__ Cout,
            const float* __restrict__ resid,
            int M, int N, int K, int act_gelu, int out_bf16,
            int inGrp, int outGrp, int rowOff, int ldc) {
  __shared__ __align__(16) unsigned short As[2][BM_ * LDSS];
  __shared__ __align__(16) unsigned short Bs[2][BM_ * LDSS];

  int tid  = threadIdx.x;
  int lane = tid & 31;
  int wv   = tid >> 5;
  int wm   = wv & 3;            // 0..3 -> 32-row strip
  int wn   = wv >> 2;           // 0..1 -> 64-col strip
  long m0  = (long)blockIdx.y * BM_;
  long n0  = (long)blockIdx.x * BN_;
  int rsel = lane & 15;
  int hs   = lane >> 4;         // 0: K 0-7/16-23, 1: K 8-15/24-31

  // stage one 128x32 A-tile and B-tile (each 512 x 16B chunks; 2 per thread)
  auto stage = [&](int buf, int k0) {
    #pragma unroll
    for (int s = 0; s < 2; ++s) {
      int ch  = tid + s * 256;          // 0..511
      int row = ch >> 2;
      int c   = ch & 3;
      const unsigned short* ga = A  + (m0 + row) * (long)K + k0 + c * 8;
      const unsigned short* gb = Bw + (n0 + row) * (long)K + k0 + c * 8;
      unsigned short* la = &As[buf][row * LDSS + c * 8];
      unsigned short* lb = &Bs[buf][row * LDSS + c * 8];
#if ASYNC_OK
      __builtin_amdgcn_global_load_async_to_lds_b128(
          (AS1 vi4_*)ga, (AS3 vi4_*)la, 0, 0);
      __builtin_amdgcn_global_load_async_to_lds_b128(
          (AS1 vi4_*)gb, (AS3 vi4_*)lb, 0, 0);
#else
      *(u32x4*)la = *(const u32x4*)ga;
      *(u32x4*)lb = *(const u32x4*)gb;
#endif
    }
  };

  v8f acc[2][4];
  #pragma unroll
  for (int i = 0; i < 2; ++i)
    #pragma unroll
    for (int j = 0; j < 4; ++j)
      acc[i][j] = (v8f){0.f, 0.f, 0.f, 0.f, 0.f, 0.f, 0.f, 0.f};

  stage(0, 0);
  int nK = K / BK_;
  for (int kt = 0; kt < nK; ++kt) {
    int buf = kt & 1;
#if ASYNC_OK
#if __has_builtin(__builtin_amdgcn_s_wait_asynccnt)
    __builtin_amdgcn_s_wait_asynccnt(0);
#else
    asm volatile("s_wait_asynccnt 0" ::: "memory");
#endif
#endif
    __syncthreads();                       // tile kt visible to all waves
    if (kt + 1 < nK) stage(buf ^ 1, (kt + 1) * BK_);

    FragAB af[2], bf[4];
    #pragma unroll
    for (int i = 0; i < 2; ++i) {
      const unsigned short* p = &As[buf][(wm * 32 + i * 16 + rsel) * LDSS + hs * 8];
      af[i].q[0] = *(const u32x4*)p;
      af[i].q[1] = *(const u32x4*)(p + 16);
    }
    #pragma unroll
    for (int j = 0; j < 4; ++j) {
      const unsigned short* p = &Bs[buf][(wn * 64 + j * 16 + rsel) * LDSS + hs * 8];
      bf[j].q[0] = *(const u32x4*)p;
      bf[j].q[1] = *(const u32x4*)(p + 16);
    }
    #pragma unroll
    for (int i = 0; i < 2; ++i)
      #pragma unroll
      for (int j = 0; j < 4; ++j)
        acc[i][j] = __builtin_amdgcn_wmma_f32_16x16x32_bf16(
            false, af[i].v, false, bf[j].v, (short)0, acc[i][j], false, false);
  }

  // epilogue
  #pragma unroll
  for (int i = 0; i < 2; ++i) {
    int mloc = wm * 32 + i * 16 + hs * 8;
    #pragma unroll
    for (int j = 0; j < 4; ++j) {
      long col = n0 + wn * 64 + j * 16 + rsel;
      float bv = bias ? bias[col] : 0.0f;
      #pragma unroll
      for (int e = 0; e < 8; ++e) {
        long m  = m0 + mloc + e;
        long mb = m / inGrp;
        long orow = mb * outGrp + rowOff + (m - mb * inGrp);
        long oidx = orow * (long)ldc + col;
        float v = acc[i][j][e] + bv;
        if (act_gelu) v = geluf(v);
        if (resid)    v += resid[oidx];
        if (out_bf16) ((unsigned short*)Cout)[oidx] = f2bf(v);
        else          ((float*)Cout)[oidx] = v;
      }
    }
  }
}

// -------------------------------- pos bias on text rows + query token insert
__global__ void k_frontend_finish(const float* __restrict__ qtok,
                                  float* __restrict__ bufx) {
  long idx = (long)blockIdx.x * blockDim.x + threadIdx.x;
  if (idx >= (long)B_ * L_ * D_) return;
  int d = (int)(idx & (D_ - 1));
  long r = idx >> 10;
  int l = (int)(r % L_);
  if (l < Q_) bufx[idx] = qtok[(long)l * D_ + d];
  else        bufx[idx] += 0.1f * (float)(l - Q_) / (float)T_;
}

// ----------------------------------------------------------- LayerNorm
__global__ void __launch_bounds__(256)
k_layernorm(const float* __restrict__ in,
            const float* __restrict__ g, const float* __restrict__ b,
            float* __restrict__ out32, unsigned short* __restrict__ out16,
            int inGrp, int outGrp) {
  int ro = blockIdx.x;
  int bb = ro / outGrp;
  int i  = ro - bb * outGrp;
  const float* row = in + ((long)bb * inGrp + i) * D_;
  __shared__ float sh1[8], sh2[8];
  float s = 0.f, ss = 0.f;
  for (int d = threadIdx.x; d < D_; d += 256) {
    float v = row[d]; s += v; ss += v * v;
  }
  #pragma unroll
  for (int off = 16; off > 0; off >>= 1) {
    s  += __shfl_xor(s,  off, 32);
    ss += __shfl_xor(ss, off, 32);
  }
  int lane = threadIdx.x & 31, wv = threadIdx.x >> 5;
  if (lane == 0) { sh1[wv] = s; sh2[wv] = ss; }
  __syncthreads();
  s = 0.f; ss = 0.f;
  #pragma unroll
  for (int w2 = 0; w2 < 8; ++w2) { s += sh1[w2]; ss += sh2[w2]; }
  float mu  = s * (1.0f / D_);
  float var = ss * (1.0f / D_) - mu * mu;
  float inv = rsqrtf(var + EPS_);
  for (int d = threadIdx.x; d < D_; d += 256) {
    float v = (row[d] - mu) * inv * g[d] + b[d];
    long o = (long)ro * D_ + d;
    if (out32) out32[o] = v;
    if (out16) out16[o] = f2bf(v);
  }
}

// ------------------------------- attention: text tokens = one-hot diagonal
__global__ void k_attn_copy_v(const float* __restrict__ qkvb,
                              float* __restrict__ o) {
  long idx = (long)blockIdx.x * blockDim.x + threadIdx.x;
  if (idx >= (long)B_ * T_ * D_) return;
  int d = (int)(idx & (D_ - 1));
  long r = idx >> 10;
  int t = (int)(r % T_);
  int b = (int)(r / T_);
  long row = (long)b * L_ + Q_ + t;
  o[row * D_ + d] = qkvb[row * (3 * D_) + 2 * D_ + d];
}

// ---------------- attention: query rows, windowed keys, online softmax
__global__ void __launch_bounds__(128)
k_attn_queries(const float* __restrict__ qkvb, float* __restrict__ o) {
  int lane = threadIdx.x & 31;
  int wave = threadIdx.x >> 5;
  int bq = blockIdx.x;
  int bb = bq / Q_;
  int qi = bq - bb * Q_;
  int start = qi * T_ / Q_ - W_;       if (start < 0)  start = 0;
  int end   = (qi + 1) * T_ / Q_ + W_; if (end > T_)   end = T_;
  const float* base = qkvb + ((long)bb * L_) * (3 * D_);
  const float* qrow = base + (long)qi * (3 * D_);

  for (int h = wave; h < H_; h += 4) {
    float qv[4];
    #pragma unroll
    for (int e = 0; e < 4; ++e) qv[e] = qrow[h * HD_ + lane * 4 + e];
    float m = -1e30f, lsum = 0.f;
    float acc[4] = {0.f, 0.f, 0.f, 0.f};
    auto process = [&](int krow) {
      const float* kh = base + (long)krow * (3 * D_) + D_     + h * HD_;
      const float* vh = base + (long)krow * (3 * D_) + 2 * D_ + h * HD_;
      float s = 0.f;
      #pragma unroll
      for (int e = 0; e < 4; ++e) s += qv[e] * kh[lane * 4 + e];
      #pragma unroll
      for (int off = 16; off > 0; off >>= 1) s += __shfl_xor(s, off, 32);
      s *= 0.08838834764831845f;                 // 1/sqrt(128)
      float mn = fmaxf(m, s);
      float sc = __expf(m - mn);
      float w  = __expf(s - mn);
      lsum = lsum * sc + w;
      m = mn;
      #pragma unroll
      for (int e = 0; e < 4; ++e) acc[e] = acc[e] * sc + w * vh[lane * 4 + e];
    };
    process(qi);                                  // diagonal (self)
    for (int j = start; j < end; ++j) process(Q_ + j);
    float inv = 1.0f / lsum;
    float* orow = o + ((long)bb * L_ + qi) * D_ + h * HD_;
    #pragma unroll
    for (int e = 0; e < 4; ++e) orow[lane * 4 + e] = acc[e] * inv;
  }
}

// ============================================================ host driver
extern "C" void kernel_launch(void* const* d_in, const int* in_sizes, int n_in,
                              void* d_out, int out_size, void* d_ws, size_t ws_size,
                              hipStream_t stream) {
  (void)in_sizes; (void)n_in; (void)out_size; (void)ws_size;

  const float* x       = (const float*)d_in[0];
  const float* conv1_w = (const float*)d_in[1];
  const float* conv1_b = (const float*)d_in[2];
  const float* conv2_w = (const float*)d_in[3];
  const float* conv2_b = (const float*)d_in[4];
  const float* qtok    = (const float*)d_in[5];
  const float* norm_g  = (const float*)d_in[30];
  const float* norm_b  = (const float*)d_in[31];
  const float* out_w   = (const float*)d_in[32];
  const float* out_b   = (const float*)d_in[33];

  // ---- workspace layout (256B aligned slices) ----
  size_t off = 0;
  auto alloc = [&](size_t bytes) -> void* {
    void* p = (char*)d_ws + off;
    off += (bytes + 255) & ~(size_t)255;
    return p;
  };
  unsigned short* conv2w16 = (unsigned short*)alloc((size_t)D_ * 3 * D_ * 2);
  unsigned short *qkvw16[NL_], *attnw16[NL_], *ff1w16[NL_], *ff2w16[NL_];
  for (int l = 0; l < NL_; ++l) {
    qkvw16[l]  = (unsigned short*)alloc((size_t)3 * D_ * D_ * 2);
    attnw16[l] = (unsigned short*)alloc((size_t)D_ * D_ * 2);
    ff1w16[l]  = (unsigned short*)alloc((size_t)4 * D_ * D_ * 2);
    ff2w16[l]  = (unsigned short*)alloc((size_t)D_ * 4 * D_ * 2);
  }
  unsigned short* outw16 = (unsigned short*)alloc((size_t)OUT_ * D_ * 2);
  float* bufx = (float*)alloc((size_t)B_ * L_ * D_ * 4);
  float* bufh = (float*)alloc((size_t)B_ * L_ * D_ * 4);
  unsigned short* hb16 = (unsigned short*)alloc((size_t)B_ * L_ * D_ * 2);
  // big arena, aliased across phases: (g1 + im2col) -> qkvbuf -> ff16
  char* arena = (char*)alloc((size_t)B_ * L_ * 3 * D_ * 4);
  float* g1            = (float*)arena;
  unsigned short* im2  = (unsigned short*)(arena + (size_t)B_ * T_ * D_ * 4);
  float* qkvbuf        = (float*)arena;
  unsigned short* ff16 = (unsigned short*)arena;

  auto cdiv = [](long a, long b) { return (unsigned)((a + b - 1) / b); };
  auto cast = [&](const float* s, unsigned short* d, long n) {
    k_cast_f32_bf16<<<cdiv(n, 256), 256, 0, stream>>>(s, d, n);
  };

  // ---- weight conversion (fp32 -> bf16) ----
  cast(conv2_w, conv2w16, (long)D_ * 3 * D_);
  for (int l = 0; l < NL_; ++l) {
    int pb = 6 + 12 * l;
    cast((const float*)d_in[pb + 2],  qkvw16[l],  (long)3 * D_ * D_);
    cast((const float*)d_in[pb + 4],  attnw16[l], (long)D_ * D_);
    cast((const float*)d_in[pb + 8],  ff1w16[l],  (long)4 * D_ * D_);
    cast((const float*)d_in[pb + 10], ff2w16[l],  (long)D_ * 4 * D_);
  }
  cast(out_w, outw16, (long)OUT_ * D_);

  // ---- conv frontend ----
  k_conv1_gelu<<<cdiv((long)B_ * T_ * D_, 256), 256, 0, stream>>>(
      x, conv1_w, conv1_b, g1);
  k_im2col_bf16<<<cdiv((long)B_ * T_ * 3 * D_, 256), 256, 0, stream>>>(g1, im2);
  {
    dim3 grid(D_ / BN_, (B_ * T_) / BM_);
    k_gemm_bf16<<<grid, 256, 0, stream>>>(
        im2, conv2w16, conv2_b, (void*)bufx, nullptr,
        B_ * T_, D_, 3 * D_, /*gelu*/1, /*bf16*/0, T_, L_, Q_, D_);
  }
  k_frontend_finish<<<cdiv((long)B_ * L_ * D_, 256), 256, 0, stream>>>(qtok, bufx);

  // ---- transformer layers ----
  for (int l = 0; l < NL_; ++l) {
    int pb = 6 + 12 * l;
    const float* ln1g = (const float*)d_in[pb + 0];
    const float* ln1b = (const float*)d_in[pb + 1];
    const float* qkvb = (const float*)d_in[pb + 3];
    const float* aob  = (const float*)d_in[pb + 5];
    const float* ln2g = (const float*)d_in[pb + 6];
    const float* ln2b = (const float*)d_in[pb + 7];
    const float* f1b  = (const float*)d_in[pb + 9];
    const float* f2b  = (const float*)d_in[pb + 11];

    k_layernorm<<<B_ * L_, 256, 0, stream>>>(bufx, ln1g, ln1b,
                                             nullptr, hb16, L_, L_);
    {
      dim3 grid((3 * D_) / BN_, (B_ * L_) / BM_);
      k_gemm_bf16<<<grid, 256, 0, stream>>>(
          hb16, qkvw16[l], qkvb, (void*)qkvbuf, nullptr,
          B_ * L_, 3 * D_, D_, 0, 0, B_ * L_, B_ * L_, 0, 3 * D_);
    }
    k_attn_copy_v<<<cdiv((long)B_ * T_ * D_, 256), 256, 0, stream>>>(qkvbuf, bufh);
    k_attn_queries<<<B_ * Q_, 128, 0, stream>>>(qkvbuf, bufh);
    cast(bufh, hb16, (long)B_ * L_ * D_);
    {
      dim3 grid(D_ / BN_, (B_ * L_) / BM_);
      k_gemm_bf16<<<grid, 256, 0, stream>>>(
          hb16, attnw16[l], aob, (void*)bufx, bufx,
          B_ * L_, D_, D_, 0, 0, B_ * L_, B_ * L_, 0, D_);
    }
    k_layernorm<<<B_ * L_, 256, 0, stream>>>(bufx, ln2g, ln2b,
                                             nullptr, hb16, L_, L_);
    {
      dim3 grid((4 * D_) / BN_, (B_ * L_) / BM_);
      k_gemm_bf16<<<grid, 256, 0, stream>>>(
          hb16, ff1w16[l], f1b, (void*)ff16, nullptr,
          B_ * L_, 4 * D_, D_, /*gelu*/1, /*bf16*/1, B_ * L_, B_ * L_, 0, 4 * D_);
    }
    {
      dim3 grid(D_ / BN_, (B_ * L_) / BM_);
      k_gemm_bf16<<<grid, 256, 0, stream>>>(
          ff16, ff2w16[l], f2b, (void*)bufx, bufx,
          B_ * L_, D_, 4 * D_, 0, 0, B_ * L_, B_ * L_, 0, D_);
    }
  }

  // ---- final LN (query rows only, compacted) + output projection ----
  k_layernorm<<<B_ * Q_, 256, 0, stream>>>(bufx, norm_g, norm_b,
                                           nullptr, hb16, L_, Q_);
  {
    dim3 grid(OUT_ / BN_, (B_ * Q_) / BM_);
    k_gemm_bf16<<<grid, 256, 0, stream>>>(
        hb16, outw16, out_b, d_out, nullptr,
        B_ * Q_, OUT_, D_, 0, 0, B_ * Q_, B_ * Q_, 0, OUT_);
  }
}
